// feed_forward_8065948582229
// MI455X (gfx1250) — compile-verified, matrix operand order checked by reading, not run on hardware
//
#include <hip/hip_runtime.h>
#include <math.h>

// ---------------------------------------------------------------------------
// Fused LayerNorm -> Linear(bf16 WMMA) -> exact GELU -> residual for MI455X.
// Shapes: x[32768, 2048] (B*S rows), W[2048, 2048], out[32768, 2048].
// GEMM: block tile 128x128, 8 wave32 waves (wave tile 32x64), K step 32,
// double-buffered LDS fed by async global->LDS loads (ASYNCcnt) when the
// toolchain exposes them.
// ---------------------------------------------------------------------------

typedef __attribute__((ext_vector_type(16))) __bf16 v16bf;
typedef __attribute__((ext_vector_type(8)))  float  v8f;
typedef __attribute__((ext_vector_type(4)))  int    v4i;

#define ROWS   32768
#define DDIM   2048
#define HDIM   2048

#define BM 128
#define BN 128
#define BK 32
#define KTILES (DDIM / BK)   // 64

#if defined(__AMDGCN__) && __has_builtin(__builtin_amdgcn_global_load_async_to_lds_b128)
#define USE_ASYNC 1
#else
#define USE_ASYNC 0
#endif

#if USE_ASYNC
typedef __attribute__((address_space(1))) v4i gv4i;   // global int4
typedef __attribute__((address_space(3))) v4i sv4i;   // LDS int4
#if __has_builtin(__builtin_amdgcn_s_wait_asynccnt)
#define WAIT_ASYNC() __builtin_amdgcn_s_wait_asynccnt(0)
#else
#define WAIT_ASYNC() asm volatile("s_wait_asynccnt 0" ::: "memory")
#endif
#else
#define WAIT_ASYNC()
#endif

union FragBF { v16bf v; uint4 u[2]; };

__device__ __forceinline__ unsigned short f2bf(float f) {
    unsigned int u = __float_as_uint(f);
    u += 0x7FFFu + ((u >> 16) & 1u);   // round-to-nearest-even
    return (unsigned short)(u >> 16);
}

// One 16-byte segment: global -> LDS (async DMA path if available).
__device__ __forceinline__ void stage_seg(const unsigned short* g, unsigned short* l) {
#if USE_ASYNC
    __builtin_amdgcn_global_load_async_to_lds_b128(
        (gv4i*)g, (sv4i*)l, /*offset=*/0, /*cpol=*/0);
#else
    *(uint4*)l = *(const uint4*)g;
#endif
}

// ---------------------------------------------------------------------------
// Kernel 1: per-row LayerNorm, write normalized activations as bf16.
// ---------------------------------------------------------------------------
__global__ __launch_bounds__(256) void ln_cast_kernel(
    const float* __restrict__ x,
    const float* __restrict__ gamma,
    const float* __restrict__ beta,
    unsigned short* __restrict__ xa)
{
    const int row = blockIdx.x;
    const int t   = threadIdx.x;
    const float* xr = x + (size_t)row * DDIM;

    float v[8];
    {
        const float4* x4 = (const float4*)xr + t * 2;
        float4 a = x4[0], b = x4[1];
        v[0]=a.x; v[1]=a.y; v[2]=a.z; v[3]=a.w;
        v[4]=b.x; v[5]=b.y; v[6]=b.z; v[7]=b.w;
    }

    float s = 0.f, ss = 0.f;
    #pragma unroll
    for (int k = 0; k < 8; ++k) { s += v[k]; ss += v[k] * v[k]; }

    #pragma unroll
    for (int off = 16; off > 0; off >>= 1) {
        s  += __shfl_xor(s,  off, 32);
        ss += __shfl_xor(ss, off, 32);
    }

    __shared__ float sS[8], sQ[8];
    const int wid = t >> 5, lid = t & 31;
    if (lid == 0) { sS[wid] = s; sQ[wid] = ss; }
    __syncthreads();

    float ts = 0.f, tq = 0.f;
    #pragma unroll
    for (int w = 0; w < 8; ++w) { ts += sS[w]; tq += sQ[w]; }

    const float mu   = ts * (1.0f / DDIM);
    const float var  = tq * (1.0f / DDIM) - mu * mu;
    const float rstd = rsqrtf(var + 1e-5f);

    unsigned short o[8];
    #pragma unroll
    for (int k = 0; k < 8; ++k) {
        const int d = t * 8 + k;
        const float xn = (v[k] - mu) * rstd * gamma[d] + beta[d];
        o[k] = f2bf(xn);
    }
    uint4 pack;
    pack.x = (unsigned)o[0] | ((unsigned)o[1] << 16);
    pack.y = (unsigned)o[2] | ((unsigned)o[3] << 16);
    pack.z = (unsigned)o[4] | ((unsigned)o[5] << 16);
    pack.w = (unsigned)o[6] | ((unsigned)o[7] << 16);
    *(uint4*)(xa + (size_t)row * DDIM + t * 8) = pack;
}

// ---------------------------------------------------------------------------
// Kernel 2: W[d][h] f32 -> Wt[h][d] bf16 (transpose + convert).
// ---------------------------------------------------------------------------
__global__ __launch_bounds__(256) void wcast_kernel(
    const float* __restrict__ W,
    unsigned short* __restrict__ Wt)
{
    __shared__ unsigned short tile[32][33];
    const int bx = blockIdx.x;          // h tile
    const int by = blockIdx.y;          // d tile
    const int tx = threadIdx.x & 31;
    const int ty = threadIdx.x >> 5;

    #pragma unroll
    for (int i = 0; i < 4; ++i) {
        const int dl = ty * 4 + i;
        tile[dl][tx] = f2bf(W[(size_t)(by * 32 + dl) * HDIM + bx * 32 + tx]);
    }
    __syncthreads();
    #pragma unroll
    for (int i = 0; i < 4; ++i) {
        const int hl = ty * 4 + i;
        Wt[(size_t)(bx * 32 + hl) * DDIM + by * 32 + tx] = tile[tx][hl];
    }
}

// ---------------------------------------------------------------------------
// Kernel 3: bf16 WMMA GEMM + bias + exact GELU + residual.
// ---------------------------------------------------------------------------
__global__ __launch_bounds__(256) void gemm_gelu_kernel(
    const unsigned short* __restrict__ A,    // [ROWS][DDIM] bf16 (normalized x)
    const unsigned short* __restrict__ Bt,   // [HDIM][DDIM] bf16 (W transposed)
    const float* __restrict__ bias,          // [HDIM]
    const float* __restrict__ xres,          // [ROWS][DDIM] f32 (residual)
    float* __restrict__ out)                 // [ROWS][HDIM] f32
{
    __shared__ unsigned short lA[2][BM * BK];   // 2 x 8 KB
    __shared__ unsigned short lB[2][BN * BK];   // 2 x 8 KB

    const int n0   = blockIdx.x * BN;   // x walks N: consecutive blocks share A panel
    const int m0   = blockIdx.y * BM;
    const int t    = threadIdx.x;
    const int wave = t >> 5;
    const int lane = t & 31;
    const int wm   = (wave & 3) << 5;   // 0,32,64,96
    const int wn   = (wave >> 2) << 6;  // 0,64
    const int lrow = lane & 15;
    const int half = lane >> 4;

    // Per-thread staging coords: 512 x 16B segments per tile, 2 per thread
    // (rows r and r+64, same 8-element column chunk).
    const int r0 = t >> 2;
    const int c0 = (t & 3) * 8;
    const unsigned short* gA = A  + (size_t)(m0 + r0) * DDIM + c0;
    const unsigned short* gB = Bt + (size_t)(n0 + r0) * DDIM + c0;
    unsigned short* dA0[2] = { &lA[0][r0 * BK + c0], &lA[1][r0 * BK + c0] };
    unsigned short* dB0[2] = { &lB[0][r0 * BK + c0], &lB[1][r0 * BK + c0] };

    v8f acc[2][4];
    #pragma unroll
    for (int i = 0; i < 2; ++i)
        #pragma unroll
        for (int j = 0; j < 4; ++j)
            acc[i][j] = (v8f){0.f,0.f,0.f,0.f,0.f,0.f,0.f,0.f};

    // Preload K tile 0 into buffer 0.
    stage_seg(gA,             dA0[0]);
    stage_seg(gA + 64 * DDIM, dA0[0] + 64 * BK);
    stage_seg(gB,             dB0[0]);
    stage_seg(gB + 64 * DDIM, dB0[0] + 64 * BK);
    WAIT_ASYNC();
    __syncthreads();

    for (int kt = 0; kt < KTILES; ++kt) {
        const int buf = kt & 1;

        // Prefetch next K tile into the other buffer while we compute.
        if (kt + 1 < KTILES) {
            const int nk = (kt + 1) * BK;
            stage_seg(gA + nk,             dA0[buf ^ 1]);
            stage_seg(gA + nk + 64 * DDIM, dA0[buf ^ 1] + 64 * BK);
            stage_seg(gB + nk,             dB0[buf ^ 1]);
            stage_seg(gB + nk + 64 * DDIM, dB0[buf ^ 1] + 64 * BK);
        }

        // Fragments per CDNA5 16-bit layout: lane-half selects K sub-ranges
        // {8h..8h+7} and {16+8h..16+8h+7}.
        FragBF fa[2], fb[4];
        #pragma unroll
        for (int i = 0; i < 2; ++i) {
            const unsigned short* pa = &lA[buf][(wm + i * 16 + lrow) * BK];
            fa[i].u[0] = *(const uint4*)(pa + half * 8);
            fa[i].u[1] = *(const uint4*)(pa + 16 + half * 8);
        }
        #pragma unroll
        for (int j = 0; j < 4; ++j) {
            const unsigned short* pb = &lB[buf][(wn + j * 16 + lrow) * BK];
            fb[j].u[0] = *(const uint4*)(pb + half * 8);
            fb[j].u[1] = *(const uint4*)(pb + 16 + half * 8);
        }

        #pragma unroll
        for (int i = 0; i < 2; ++i)
            #pragma unroll
            for (int j = 0; j < 4; ++j)
                acc[i][j] = __builtin_amdgcn_wmma_f32_16x16x32_bf16(
                    false, fa[i].v, false, fb[j].v,
                    (short)0, acc[i][j], false, false);

        if (kt + 1 < KTILES) { WAIT_ASYNC(); }
        __syncthreads();
    }

    // Epilogue: bias + exact GELU + residual.
    // C/D layout: lane&15 -> N, vgpr r -> M = r + 8*(lane>>4).
    #pragma unroll
    for (int i = 0; i < 2; ++i) {
        #pragma unroll
        for (int j = 0; j < 4; ++j) {
            const int colg = n0 + wn + j * 16 + lrow;
            const float bb = bias[colg];
            #pragma unroll
            for (int r = 0; r < 8; ++r) {
                const int rowg = m0 + wm + i * 16 + r + half * 8;
                const size_t off = (size_t)rowg * HDIM + colg;
                float h = acc[i][j][r] + bb;
                h = 0.5f * h * (1.0f + erff(h * 0.70710678118654752f));
                out[off] = h + xres[off];
            }
        }
    }
}

// ---------------------------------------------------------------------------
extern "C" void kernel_launch(void* const* d_in, const int* in_sizes, int n_in,
                              void* d_out, int out_size, void* d_ws, size_t ws_size,
                              hipStream_t stream) {
    const float* x     = (const float*)d_in[0];   // [16,2048,2048]
    const float* gamma = (const float*)d_in[1];   // [2048]
    const float* beta  = (const float*)d_in[2];   // [2048]
    const float* W     = (const float*)d_in[3];   // [2048,2048]
    const float* bias  = (const float*)d_in[4];   // [2048]
    float* out = (float*)d_out;

    unsigned short* Abf = (unsigned short*)d_ws;                 // 128 MB
    unsigned short* Wt  = Abf + (size_t)ROWS * DDIM;             // + 8 MB

    ln_cast_kernel<<<ROWS, 256, 0, stream>>>(x, gamma, beta, Abf);
    wcast_kernel<<<dim3(HDIM / 32, DDIM / 32), 256, 0, stream>>>(W, Wt);
    gemm_gelu_kernel<<<dim3(HDIM / BN, ROWS / BM), 256, 0, stream>>>(
        Abf, Wt, bias, x, out);
}
